// M1_65893388255551
// MI455X (gfx1250) — compile-verified, matrix operand order checked by reading, not run on hardware
//
#include <hip/hip_runtime.h>
#include <hip/hip_bf16.h>
#include <math.h>

// ---------------- model constants ----------------
#define BB   256      // batch
#define KK   16       // neighbours
#define OBS  50       // sequence length
#define HID  256
#define EMB  128
#define UEMB 64
#define HALF 64
#define POI  30000
#define NH_MAIN 8
#define NH_SOC  2

typedef __attribute__((ext_vector_type(16))) _Float16 v16h;
typedef __attribute__((ext_vector_type(8)))  float    v8f;

// ---------------- device helpers ----------------
__device__ __forceinline__ float selu_f(float x) {
    const float a  = 1.6732632423543772f;
    const float sc = 1.0507009873554805f;
    return sc * (x > 0.f ? x : a * (__expf(x) - 1.f));
}

// wait for all outstanding async (global<->LDS) transfers of this wave
__device__ __forceinline__ void wait_async0() {
#if __has_builtin(__builtin_amdgcn_s_wait_asynccnt)
    __builtin_amdgcn_s_wait_asynccnt(0);
#else
    asm volatile("s_wait_asynccnt 0" ::: "memory");
#endif
}

// issue a 32-byte (2 x B128) async copy global -> LDS.
// lds  : wave-relative LDS byte address (low 32 bits of the generic pointer)
// gofs : per-lane byte offset from base (GVS mode: SGPR64 base + VGPR u32)
__device__ __forceinline__ void async_copy32(unsigned lds, unsigned gofs,
                                             const void* base) {
    asm volatile("global_load_async_to_lds_b128 %0, %1, %2"
                 :: "v"(lds), "v"(gofs), "s"(base) : "memory");
    asm volatile("global_load_async_to_lds_b128 %0, %1, %2 offset:16"
                 :: "v"(lds), "v"(gofs), "s"(base) : "memory");
}

// =====================================================================
// GEMM: C[M,N] = act(A[M,K] @ W[N,K]^T + bias[N])
// A,W f16 row-major, K % 32 == 0. 128 threads = 4 waves.
// Block tile 64x64x32; double-buffered LDS tiles filled by
// GLOBAL_LOAD_ASYNC_TO_LDS_B128 (ASYNCcnt) overlapping the WMMA stream.
// Each wave computes a 16x64 strip: 4 x v_wmma_f32_16x16x32_f16 per k-step.
// =====================================================================
#define BM 64
#define BN 64
#define BKK 32
#define LDA 40   // padded LDS row stride (halves)

template<int ACT>
__launch_bounds__(128)
__global__ void gemm_wmma(const _Float16* __restrict__ A,
                          const _Float16* __restrict__ W,
                          const float* __restrict__ bias,
                          float* __restrict__ Cf,
                          _Float16* __restrict__ Ch,
                          int M, int N, int K)
{
    __shared__ _Float16 As[2][BM][LDA];
    __shared__ _Float16 Bs[2][BN][LDA];

    const int tid   = threadIdx.x;
    const int wave  = tid >> 5;
    const int lane  = tid & 31;
    const int hlf   = lane >> 4;      // 0/1
    const int l15   = lane & 15;
    const int m0    = blockIdx.y * BM;
    const int n0    = blockIdx.x * BN;

    const int ldr = tid >> 1;         // 0..63 (tile row this thread stages)
    const int ldc = (tid & 1) * 16;   // 0 or 16 (halves)

    const int  gm   = m0 + ldr;
    const int  gn   = n0 + ldr;
    const bool am_ok = gm < M;
    const bool bn_ok = gn < N;

    // wave-relative LDS byte addresses of this thread's staging slots
    const unsigned ldsA0 = (unsigned)(size_t)&As[0][ldr][ldc];
    const unsigned ldsA1 = (unsigned)(size_t)&As[1][ldr][ldc];
    const unsigned ldsB0 = (unsigned)(size_t)&Bs[0][ldr][ldc];
    const unsigned ldsB1 = (unsigned)(size_t)&Bs[1][ldr][ldc];

    // zero-fill out-of-range rows once; their async issues are skipped below,
    // so the zeros persist for the whole k-loop.
    {
        const uint4 z = make_uint4(0u,0u,0u,0u);
        if (!am_ok) {
            *reinterpret_cast<uint4*>(&As[0][ldr][ldc]) = z;
            *reinterpret_cast<uint4*>(&As[0][ldr][ldc + 8]) = z;
            *reinterpret_cast<uint4*>(&As[1][ldr][ldc]) = z;
            *reinterpret_cast<uint4*>(&As[1][ldr][ldc + 8]) = z;
        }
        if (!bn_ok) {
            *reinterpret_cast<uint4*>(&Bs[0][ldr][ldc]) = z;
            *reinterpret_cast<uint4*>(&Bs[0][ldr][ldc + 8]) = z;
            *reinterpret_cast<uint4*>(&Bs[1][ldr][ldc]) = z;
            *reinterpret_cast<uint4*>(&Bs[1][ldr][ldc + 8]) = z;
        }
    }
    __syncthreads();

    v8f acc[4];
    #pragma unroll
    for (int t = 0; t < 4; ++t)
        #pragma unroll
        for (int r = 0; r < 8; ++r) acc[t][r] = 0.f;

    const unsigned aRow = (unsigned)((size_t)gm * K + ldc) * 2u; // byte offsets
    const unsigned bRow = (unsigned)((size_t)gn * K + ldc) * 2u;

    // prologue: stage tile 0 into buffer 0
    if (am_ok) async_copy32(ldsA0, aRow, A);
    if (bn_ok) async_copy32(ldsB0, bRow, W);
    wait_async0();
    __syncthreads();

    const int nk = K / BKK;
    for (int kt = 0; kt < nk; ++kt) {
        const int cur = kt & 1;

        // issue async fill of the other buffer for tile kt+1 (overlaps WMMAs)
        if (kt + 1 < nk) {
            const unsigned kb = (unsigned)(kt + 1) * (BKK * 2u);
            if (am_ok) async_copy32(cur ? ldsA0 : ldsA1, aRow + kb, A);
            if (bn_ok) async_copy32(cur ? ldsB0 : ldsB1, bRow + kb, W);
        }

        // ---- A fragment for this wave (rows wave*16 .. +15) ----
        union { v16h v; float f[8]; } af;
        {
            const float* ar = reinterpret_cast<const float*>(&As[cur][wave * 16 + l15][0]);
            const int h4 = hlf * 4;
            #pragma unroll
            for (int v = 0; v < 4; ++v) { af.f[v] = ar[h4 + v]; af.f[4 + v] = ar[8 + h4 + v]; }
        }
        // ---- 4 column tiles ----
        #pragma unroll
        for (int t = 0; t < 4; ++t) {
            union { v16h v; float f[8]; } bf;
            const float* br = reinterpret_cast<const float*>(&Bs[cur][t * 16 + l15][0]);
            const int h4 = hlf * 4;
            #pragma unroll
            for (int v = 0; v < 4; ++v) { bf.f[v] = br[h4 + v]; bf.f[4 + v] = br[8 + h4 + v]; }
            acc[t] = __builtin_amdgcn_wmma_f32_16x16x32_f16(
                false, af.v, false, bf.v, (short)0, acc[t], false, false);
        }

        // my async writes done + everyone finished reading/writing this round
        wait_async0();
        __syncthreads();
    }

    // ---- epilogue: bias + activation + store ----
    #pragma unroll
    for (int t = 0; t < 4; ++t) {
        int col = n0 + t * 16 + l15;
        if (col >= N) continue;
        float bv = bias ? bias[col] : 0.f;
        #pragma unroll
        for (int r = 0; r < 8; ++r) {
            int row = m0 + wave * 16 + hlf * 8 + r;
            if (row >= M) continue;
            float v = acc[t][r] + bv;
            if (ACT == 1) v = v > 0.f ? v : 0.f;
            else if (ACT == 2) v = selu_f(v);
            size_t off = (size_t)row * N + col;
            if (Cf) Cf[off] = v;
            if (Ch) Ch[off] = (_Float16)v;
        }
    }
}

// =====================================================================
// Fused MHA: qkv[R,3E] (f16) -> ctx[R,E] (f16). One block per (seq, head).
// =====================================================================
__launch_bounds__(64)
__global__ void attention_kernel(const _Float16* __restrict__ qkv,
                                 _Float16* __restrict__ ctx,
                                 int S, int E, int H)
{
    extern __shared__ _Float16 sh[];
    const int D   = E / H;
    const int seq = blockIdx.x;
    const int h   = blockIdx.y;
    _Float16* Qs = sh;
    _Float16* Ks = sh + S * D;
    _Float16* Vs = sh + 2 * S * D;

    for (int idx = threadIdx.x; idx < S * D; idx += blockDim.x) {
        int t = idx / D, dd = idx % D;
        size_t ro = ((size_t)(seq * S + t)) * (3 * E) + h * D + dd;
        Qs[idx] = qkv[ro];
        Ks[idx] = qkv[ro + E];
        Vs[idx] = qkv[ro + 2 * E];
    }
    __syncthreads();

    int t = threadIdx.x;
    if (t < S) {
        const float scale = rsqrtf((float)D);
        float s[64];
        float mx = -1e30f;
        for (int j = 0; j < S; ++j) {
            float dot = 0.f;
            for (int dd = 0; dd < D; ++dd)
                dot += (float)Qs[t * D + dd] * (float)Ks[j * D + dd];
            s[j] = dot * scale;
            mx = fmaxf(mx, s[j]);
        }
        float sum = 0.f;
        for (int j = 0; j < S; ++j) { s[j] = __expf(s[j] - mx); sum += s[j]; }
        float inv = 1.f / sum;
        for (int dd = 0; dd < D; ++dd) {
            float o = 0.f;
            for (int j = 0; j < S; ++j) o += s[j] * (float)Vs[j * D + dd];
            ctx[((size_t)(seq * S + t)) * E + h * D + dd] = (_Float16)(o * inv);
        }
    }
}

// =====================================================================
// LayerNorm(x + y) with affine; writes f32 (for next residual) + f16.
// =====================================================================
__launch_bounds__(256)
__global__ void ln_residual_kernel(const float* __restrict__ x,
                                   const float* __restrict__ y,
                                   const float* __restrict__ g,
                                   const float* __restrict__ b,
                                   float* __restrict__ outf,
                                   _Float16* __restrict__ outh,
                                   int E)
{
    __shared__ float red[256];
    const int row = blockIdx.x, c = threadIdx.x;
    const size_t off = (size_t)row * E + c;
    float v = x[off] + y[off];
    red[c] = v; __syncthreads();
    for (int s = 128; s > 0; s >>= 1) { if (c < s) red[c] += red[c + s]; __syncthreads(); }
    float mean = red[0] / (float)E; __syncthreads();
    float d = v - mean;
    red[c] = d * d; __syncthreads();
    for (int s = 128; s > 0; s >>= 1) { if (c < s) red[c] += red[c + s]; __syncthreads(); }
    float var = red[0] / (float)E;
    float o = d * rsqrtf(var + 1e-5f) * g[c] + b[c];
    outf[off] = o;
    outh[off] = (_Float16)o;
}

// =====================================================================
// token embedding gather + positional encoding
// =====================================================================
__launch_bounds__(256)
__global__ void embed_tokens_kernel(const int* __restrict__ poi_id,
                                    const int* __restrict__ sem_id,
                                    const int* __restrict__ tid_id,
                                    const float* __restrict__ poi_emb,
                                    const float* __restrict__ sem_emb,
                                    const float* __restrict__ tid_emb,
                                    float* __restrict__ xf,
                                    _Float16* __restrict__ xh,
                                    int ntok, int S, int row_off)
{
    const int tok = blockIdx.x;
    const int c   = threadIdx.x;
    if (tok >= ntok) return;
    float v;
    if (c < EMB)              v = poi_emb[(size_t)poi_id[tok] * EMB + c];
    else if (c < EMB + HALF)  v = sem_emb[(size_t)sem_id[tok] * HALF + (c - EMB)];
    else                      v = tid_emb[(size_t)tid_id[tok] * HALF + (c - EMB - HALF)];
    const int s = tok % S;
    float ang = (float)s * __expf(-(float)(c & ~1) * (logf(10000.f) / (float)HID));
    v += (c & 1) ? __cosf(ang) : __sinf(ang);
    size_t off = (size_t)(row_off + tok) * HID + c;
    xf[off] = v;
    xh[off] = (_Float16)v;
}

__launch_bounds__(256)
__global__ void embed_loc_kernel(const float* __restrict__ loc,
                                 const float* __restrict__ loc_w,
                                 const float* __restrict__ loc_b,
                                 float* __restrict__ xf,
                                 _Float16* __restrict__ xh,
                                 int ntok, int S)
{
    const int tok = blockIdx.x;
    const int c   = threadIdx.x;
    if (tok >= ntok) return;
    float v = loc_w[c * 2 + 0] * loc[(size_t)tok * 2 + 0]
            + loc_w[c * 2 + 1] * loc[(size_t)tok * 2 + 1]
            + loc_b[c];
    const int s = tok % S;
    float ang = (float)s * __expf(-(float)(c & ~1) * (logf(10000.f) / (float)HID));
    v += (c & 1) ? __cosf(ang) : __sinf(ang);
    size_t off = (size_t)tok * HID + c;
    xf[off] = v;
    xh[off] = (_Float16)v;
}

// =====================================================================
// misc gather / copy kernels
// =====================================================================
__global__ void f32_to_f16_kernel(const float* __restrict__ in,
                                  _Float16* __restrict__ out, long n)
{
    long i = (long)blockIdx.x * blockDim.x + threadIdx.x;
    long st = (long)gridDim.x * blockDim.x;
    for (; i < n; i += st) out[i] = (_Float16)in[i];
}

// soc_seq[b, j] = (j<K) ? last-token of social seq (B + b*K + j) : last-token of main seq b
__launch_bounds__(256)
__global__ void build_sseq_kernel(const float* __restrict__ xf,
                                  float* __restrict__ sf,
                                  _Float16* __restrict__ shh)
{
    const int row = blockIdx.x;          // 0 .. B*(K+1)-1
    const int c   = threadIdx.x;
    const int b   = row / (KK + 1);
    const int j   = row % (KK + 1);
    const int seq = (j < KK) ? (BB + b * KK + j) : b;
    float v = xf[((size_t)seq * OBS + (OBS - 1)) * HID + c];
    size_t off = (size_t)row * HID + c;
    sf[off] = v;
    shh[off] = (_Float16)v;
}

__launch_bounds__(256)
__global__ void merge_social_kernel(const float* __restrict__ sf, _Float16* __restrict__ mg)
{
    const int b = blockIdx.x, c = threadIdx.x;
    mg[(size_t)b * 576 + c] =
        (_Float16)sf[((size_t)(b * (KK + 1) + KK)) * HID + c];
}

__launch_bounds__(64)
__global__ void merge_user_kernel(const int* __restrict__ uid,
                                  const float* __restrict__ uemb,
                                  _Float16* __restrict__ mg)
{
    const int b = blockIdx.x, c = threadIdx.x;
    mg[(size_t)b * 576 + 256 + c] = (_Float16)uemb[(size_t)uid[b] * UEMB + c];
}

__launch_bounds__(256)
__global__ void merge_loc_kernel(const float* __restrict__ xf, _Float16* __restrict__ mg)
{
    const int b = blockIdx.x, c = threadIdx.x;
    mg[(size_t)b * 576 + 320 + c] =
        (_Float16)xf[((size_t)b * OBS + (OBS - 1)) * HID + c];
}

// row-wise log_softmax over N columns
__launch_bounds__(256)
__global__ void logsoftmax_kernel(const float* __restrict__ in,
                                  float* __restrict__ out, int N)
{
    __shared__ float red[256];
    const int row = blockIdx.x, tid = threadIdx.x;
    const float* r = in + (size_t)row * N;
    float mx = -1e30f;
    for (int i = tid; i < N; i += 256) mx = fmaxf(mx, r[i]);
    red[tid] = mx; __syncthreads();
    for (int s = 128; s > 0; s >>= 1) { if (tid < s) red[tid] = fmaxf(red[tid], red[tid + s]); __syncthreads(); }
    mx = red[0]; __syncthreads();
    float sum = 0.f;
    for (int i = tid; i < N; i += 256) sum += __expf(r[i] - mx);
    red[tid] = sum; __syncthreads();
    for (int s = 128; s > 0; s >>= 1) { if (tid < s) red[tid] += red[tid + s]; __syncthreads(); }
    const float lse = mx + logf(red[0]);
    for (int i = tid; i < N; i += 256) out[(size_t)row * N + i] = r[i] - lse;
}

// =====================================================================
// host-side orchestration
// =====================================================================
static inline char* bump(char*& p, size_t bytes) {
    char* r = p;
    p += (bytes + 255) & ~(size_t)255;
    return r;
}

static void conv16(hipStream_t st, const void* in, _Float16* out, long n) {
    f32_to_f16_kernel<<<dim3(1024), dim3(256), 0, st>>>((const float*)in, out, n);
}

// one post-norm transformer encoder layer on xf/xh (R = nseq*S rows, E=256)
static void enc_layer(hipStream_t st, float* xf, _Float16* xh,
                      const _Float16* qkv_w, const float* qkv_b,
                      const _Float16* out_w, const float* out_b,
                      const float* ln1_g, const float* ln1_b,
                      const _Float16* w1, const float* b1,
                      const _Float16* w2, const float* b2,
                      const float* ln2_g, const float* ln2_b,
                      int nseq, int S, int H,
                      _Float16* qkvb, _Float16* ctxb, _Float16* ffnb, float* yb)
{
    const int R = nseq * S;
    const dim3 blk(128);
    const int gy = (R + 63) / 64;
    gemm_wmma<0><<<dim3((3 * HID + 63) / 64, gy), blk, 0, st>>>(
        xh, qkv_w, qkv_b, nullptr, qkvb, R, 3 * HID, HID);
    const int D = HID / H;
    attention_kernel<<<dim3(nseq, H), dim3(64), 3 * S * D * sizeof(_Float16), st>>>(
        qkvb, ctxb, S, HID, H);
    gemm_wmma<0><<<dim3(HID / 64, gy), blk, 0, st>>>(
        ctxb, out_w, out_b, yb, nullptr, R, HID, HID);
    ln_residual_kernel<<<dim3(R), dim3(256), 0, st>>>(xf, yb, ln1_g, ln1_b, xf, xh, HID);
    gemm_wmma<1><<<dim3(2 * HID / 64, gy), blk, 0, st>>>(
        xh, w1, b1, nullptr, ffnb, R, 2 * HID, HID);
    gemm_wmma<0><<<dim3(HID / 64, gy), blk, 0, st>>>(
        ffnb, w2, b2, yb, nullptr, R, HID, 2 * HID);
    ln_residual_kernel<<<dim3(R), dim3(256), 0, st>>>(xf, yb, ln2_g, ln2_b, xf, xh, HID);
}

// d_in index map (setup_inputs() dict order, params flattened in insertion order)
enum {
    IN_X = 0, IN_USER, IN_SEM, IN_TID, IN_LOC, IN_SOC, IN_SOC_TID, IN_SOC_SEM,
    P_POI_EMB, P_USER_EMB, P_SEM_EMB, P_TID_EMB, P_LOC_W, P_LOC_B,
    PT_BASE = 14,   // t : qkv_w,qkv_b,out_w,out_b,ln1_g,ln1_b,w1,b1,w2,b2,ln2_g,ln2_b
    PL_BASE = 26,   // tl
    PS_BASE = 38,   // ts
    P_FC_W = 50, P_FC_B, P_FCS_W, P_FCS_B, P_FCL_W, P_FCL_B
};
enum { W_QKV_W = 0, W_QKV_B, W_OUT_W, W_OUT_B, W_LN1_G, W_LN1_B,
       W_W1, W_B1, W_W2, W_B2, W_LN2_G, W_LN2_B };

extern "C" void kernel_launch(void* const* d_in, const int* in_sizes, int n_in,
                              void* d_out, int out_size, void* d_ws, size_t ws_size,
                              hipStream_t stream)
{
    (void)in_sizes; (void)n_in; (void)out_size; (void)ws_size;
    char* ws = (char*)d_ws;

    const int NSEQ1 = BB + BB * KK;        // 4352 sequences through shared encoder
    const int R1    = NSEQ1 * OBS;         // 217600 token rows
    const int R3    = BB * (KK + 1);       // 4352 rows for social-mix encoder
    const long LAYER_W = (long)(3 * HID + HID + 2 * HID + 2 * HID) * HID; // f16 elems per layer

    // ---- workspace carve-out ----
    float*     xa_f   = (float*)    bump(ws, (size_t)R1 * HID * 4);
    _Float16*  xa_h   = (_Float16*) bump(ws, (size_t)R1 * HID * 2);
    float*     y_f    = (float*)    bump(ws, (size_t)R1 * HID * 4);  // also reused for fcs out
    _Float16*  qkv_b_ = (_Float16*) bump(ws, (size_t)R1 * 3 * HID * 2);
    _Float16*  ctx_b  = (_Float16*) bump(ws, (size_t)R1 * HID * 2);
    _Float16*  ffn_b  = (_Float16*) bump(ws, (size_t)R1 * 2 * HID * 2);
    float*     ss_f   = (float*)    bump(ws, (size_t)R3 * HID * 4);
    _Float16*  ss_h   = (_Float16*) bump(ws, (size_t)R3 * HID * 2);
    _Float16*  merge  = (_Float16*) bump(ws, (size_t)BB * 576 * 2);
    _Float16*  pred_h = (_Float16*) bump(ws, (size_t)BB * HID * 2);

    _Float16* tW  = (_Float16*)bump(ws, (size_t)6 * LAYER_W * 2);
    _Float16* tlW = (_Float16*)bump(ws, (size_t)6 * LAYER_W * 2);
    _Float16* tsW = (_Float16*)bump(ws, (size_t)2 * LAYER_W * 2);
    _Float16* fcW  = (_Float16*)bump(ws, (size_t)HID * 576 * 2);
    _Float16* fcsW = (_Float16*)bump(ws, (size_t)POI * HID * 2);
    _Float16* fclW = (_Float16*)bump(ws, (size_t)2 * HID * 2);

    // ---- convert weights fp32 -> fp16 (once per call; L2-resident afterwards) ----
    // layout inside tW: per layer l: [qkv_w | out_w | w1 | w2]
    auto conv_enc = [&](int base, _Float16* dst, int L) {
        const long sq = (long)3 * HID * HID, so = (long)HID * HID;
        const long s1 = (long)2 * HID * HID, s2 = (long)2 * HID * HID;
        for (int l = 0; l < L; ++l) {
            _Float16* p = dst + (long)l * LAYER_W;
            conv16(stream, (const float*)d_in[base + W_QKV_W] + l * sq, p, sq);             p += sq;
            conv16(stream, (const float*)d_in[base + W_OUT_W] + l * so, p, so);             p += so;
            conv16(stream, (const float*)d_in[base + W_W1]    + l * s1, p, s1);             p += s1;
            conv16(stream, (const float*)d_in[base + W_W2]    + l * s2, p, s2);
        }
    };
    conv_enc(PT_BASE, tW, 6);
    conv_enc(PL_BASE, tlW, 6);
    conv_enc(PS_BASE, tsW, 2);
    conv16(stream, d_in[P_FC_W],  fcW,  (long)HID * 576);
    conv16(stream, d_in[P_FCS_W], fcsW, (long)POI * HID);
    conv16(stream, d_in[P_FCL_W], fclW, (long)2 * HID);

    auto run_encoder = [&](float* xf, _Float16* xh, const _Float16* Wbase, int base,
                           int L, int nseq, int S, int H) {
        const long sq = (long)3 * HID * HID, so = (long)HID * HID;
        const long s1 = (long)2 * HID * HID;
        for (int l = 0; l < L; ++l) {
            const _Float16* p = Wbase + (long)l * LAYER_W;
            enc_layer(stream, xf, xh,
                      p,                (const float*)d_in[base + W_QKV_B] + l * 3 * HID,
                      p + sq,           (const float*)d_in[base + W_OUT_B] + l * HID,
                      (const float*)d_in[base + W_LN1_G] + l * HID,
                      (const float*)d_in[base + W_LN1_B] + l * HID,
                      p + sq + so,      (const float*)d_in[base + W_B1] + l * 2 * HID,
                      p + sq + so + s1, (const float*)d_in[base + W_B2] + l * HID,
                      (const float*)d_in[base + W_LN2_G] + l * HID,
                      (const float*)d_in[base + W_LN2_B] + l * HID,
                      nseq, S, H, qkv_b_, ctx_b, ffn_b, y_f);
        }
    };

    // ---- 1. embeddings + PE: main seqs (rows 0..B*S) then social seqs ----
    embed_tokens_kernel<<<dim3(BB * OBS), dim3(256), 0, stream>>>(
        (const int*)d_in[IN_X], (const int*)d_in[IN_SEM], (const int*)d_in[IN_TID],
        (const float*)d_in[P_POI_EMB], (const float*)d_in[P_SEM_EMB], (const float*)d_in[P_TID_EMB],
        xa_f, xa_h, BB * OBS, OBS, 0);
    embed_tokens_kernel<<<dim3(BB * KK * OBS), dim3(256), 0, stream>>>(
        (const int*)d_in[IN_SOC], (const int*)d_in[IN_SOC_SEM], (const int*)d_in[IN_SOC_TID],
        (const float*)d_in[P_POI_EMB], (const float*)d_in[P_SEM_EMB], (const float*)d_in[P_TID_EMB],
        xa_f, xa_h, BB * KK * OBS, OBS, BB * OBS);

    // ---- 2. shared 6-layer encoder over main + social batch ----
    run_encoder(xa_f, xa_h, tW, PT_BASE, 6, NSEQ1, OBS, NH_MAIN);

    // ---- 3. build social sequence [B, K+1, HID] from last tokens ----
    build_sseq_kernel<<<dim3(R3), dim3(256), 0, stream>>>(xa_f, ss_f, ss_h);

    // ---- 4. 2-layer social-mix encoder (no PE), S=17, H=2 ----
    run_encoder(ss_f, ss_h, tsW, PS_BASE, 2, BB, KK + 1, NH_SOC);
    merge_social_kernel<<<dim3(BB), dim3(256), 0, stream>>>(ss_f, merge);

    // ---- 5. loc branch: embed + 6-layer encoder (reuses big scratch) ----
    embed_loc_kernel<<<dim3(BB * OBS), dim3(256), 0, stream>>>(
        (const float*)d_in[IN_LOC], (const float*)d_in[P_LOC_W], (const float*)d_in[P_LOC_B],
        xa_f, xa_h, BB * OBS, OBS);
    run_encoder(xa_f, xa_h, tlW, PL_BASE, 6, BB, OBS, NH_MAIN);
    merge_loc_kernel<<<dim3(BB), dim3(256), 0, stream>>>(xa_f, merge);

    // ---- 6. user embedding into merge ----
    merge_user_kernel<<<dim3(BB), dim3(64), 0, stream>>>(
        (const int*)d_in[IN_USER], (const float*)d_in[P_USER_EMB], merge);

    // ---- 7. pred = merge @ fc_w^T + fc_b   [B,576] -> [B,256] ----
    gemm_wmma<0><<<dim3(HID / 64, (BB + 63) / 64), dim3(128), 0, stream>>>(
        merge, fcW, (const float*)d_in[P_FC_B], nullptr, pred_h, BB, HID, 576);

    // ---- 8. score = log_softmax(selu(pred @ fcs_w^T + fcs_b)) ----
    gemm_wmma<2><<<dim3((POI + 63) / 64, (BB + 63) / 64), dim3(128), 0, stream>>>(
        pred_h, fcsW, (const float*)d_in[P_FCS_B], y_f, nullptr, BB, POI, HID);
    logsoftmax_kernel<<<dim3(BB), dim3(256), 0, stream>>>(y_f, (float*)d_out, POI);

    // ---- 9. pred_loc = pred @ fcl_w^T + fcl_b  -> tail of d_out ----
    gemm_wmma<0><<<dim3(1, (BB + 63) / 64), dim3(128), 0, stream>>>(
        pred_h, fclW, (const float*)d_in[P_FCL_B],
        (float*)d_out + (size_t)BB * POI, nullptr, BB, 2, HID);
}
